// SiamFC_HEAD_dweasy_87625922773188
// MI455X (gfx1250) — matrix operand order repaired and checked by reading, not compiled, not run on hardware
//
#include <hip/hip_runtime.h>
#include <hip/hip_bf16.h>
#include <math.h>

// ---- Problem constants (SiamFC head) ----
// z: (64,64,16,16) f32, x: (64,64,96,96) f32, wz/wx: (64,1,3,3), bz/bx: (64,)
// out: (64,1,97,97) f32, scale 0.001
#define NB   64     // batch
#define NC   64     // channels
#define XH   96
#define XW   96
#define ZH   16
#define ZW   16
#define SPAD 128    // padded column count for XT (x cols live at [8,104))
#define OW   97
#define OH   97
#define OXT  7      // ceil(97/16) ox tiles

typedef _Float16 h8   __attribute__((ext_vector_type(8)));
typedef _Float16 h16  __attribute__((ext_vector_type(16)));
typedef float    f8   __attribute__((ext_vector_type(8)));

__device__ __forceinline__ float gelu_exact(float u) {
    return 0.5f * u * (1.0f + erff(u * 0.70710678118654752440f));
}

// ---------- zero the output (harness poisons it with 0xAA) ----------
__global__ void k_zero_out(float* __restrict__ out, int n) {
    int i = blockIdx.x * blockDim.x + threadIdx.x;
    if (i < n) out[i] = 0.0f;
}

// ---------- prelude on x: depthwise3x3+bias, residual, exact GELU ----------
// Writes f16 transposed layout XT[b][row][s(0..127)][c], x data at s = col+8,
// zeros in the pad columns so the correlate kernel needs no horizontal checks.
__global__ void k_prep_x(const float* __restrict__ x, const float* __restrict__ wx,
                         const float* __restrict__ bx, _Float16* __restrict__ XT) {
    long id = (long)blockIdx.x * blockDim.x + threadIdx.x;
    const long total = (long)NB * XH * SPAD * NC;
    if (id >= total) return;
    int c   = (int)(id & (NC - 1));
    long t  = id >> 6;
    int s   = (int)(t & (SPAD - 1)); t >>= 7;
    int row = (int)(t % XH);
    int b   = (int)(t / XH);

    _Float16 outv = (_Float16)0.0f;
    int col = s - 8;
    if (col >= 0 && col < XW) {
        const float* xc = x + ((long)(b * NC + c) * XH) * XW;
        float acc = bx[c];
        #pragma unroll
        for (int dr = 0; dr < 3; ++dr) {
            int rr = row + dr - 1;
            if (rr < 0 || rr >= XH) continue;
            #pragma unroll
            for (int dc = 0; dc < 3; ++dc) {
                int cc = col + dc - 1;
                if (cc < 0 || cc >= XW) continue;
                acc += wx[(c * 3 + dr) * 3 + dc] * xc[rr * XW + cc];
            }
        }
        outv = (_Float16)gelu_exact(xc[row * XW + col] + acc);
    }
    XT[id] = outv;   // id == (((b*XH+row)*SPAD + s)*NC + c)
}

// ---------- prelude on z: writes ZT[b][j][i][c] (channel-innermost) ----------
__global__ void k_prep_z(const float* __restrict__ z, const float* __restrict__ wz,
                         const float* __restrict__ bz, _Float16* __restrict__ ZT) {
    long id = (long)blockIdx.x * blockDim.x + threadIdx.x;
    const long total = (long)NB * ZW * ZH * NC;
    if (id >= total) return;
    int c = (int)(id & (NC - 1));
    int i = (int)((id >> 6) & 15);
    int j = (int)((id >> 10) & 15);
    int b = (int)(id >> 14);

    const float* zc = z + ((long)(b * NC + c) * ZH) * ZW;
    float acc = bz[c];
    #pragma unroll
    for (int dr = 0; dr < 3; ++dr) {
        int rr = i + dr - 1;
        if (rr < 0 || rr >= ZH) continue;
        #pragma unroll
        for (int dc = 0; dc < 3; ++dc) {
            int cc = j + dc - 1;
            if (cc < 0 || cc >= ZW) continue;
            acc += wz[(c * 3 + dr) * 3 + dc] * zc[rr * ZW + cc];
        }
    }
    ZT[id] = (_Float16)gelu_exact(zc[i * ZW + j] + acc);
}

// ---------- WMMA correlation ----------
// One wave per (b, y', ox_tile). GEMM tile: M=16 output cols, N=16 template
// rows i, K=1024 over (j outer, c inner). D[m,i] scattered to out[y'-i+8, ox].
__global__ void __launch_bounds__(256)
k_corr(const _Float16* __restrict__ XT, const _Float16* __restrict__ ZT,
       float* __restrict__ out) {
    int wave = blockIdx.x * (blockDim.x >> 5) + (threadIdx.x >> 5);
    int lane = threadIdx.x & 31;
    int ot  = wave % OXT;
    int t   = wave / OXT;
    int yp  = t % XH;          // source row y' in 0..95
    int b   = t / XH;

    int hi  = lane >> 4;       // lane half selects K sub-ranges
    int m   = lane & 15;       // A: row M ; B: column N (= template row i)
    int oxb = ot * 16;

    const _Float16* xrow = XT + (long)(b * XH + yp) * SPAD * NC;
    const _Float16* zb   = ZT + (long)b * ZW * ZH * NC;

    f8 acc = {};
    for (int j0 = 0; j0 < ZW; ++j0) {
        // A: XT[b][y'][oxb+m+j0][c]  — two aligned 16B chunks per lane
        const _Float16* pa = xrow + (long)(oxb + m + j0) * NC;
        // B: ZT[b][j0][i=m][c]
        const _Float16* pb = zb + (long)(j0 * ZH + m) * NC;
        __builtin_prefetch(pa + NC, 0, 1);   // next j0's sliding-window column
        #pragma unroll
        for (int c0 = 0; c0 < NC; c0 += 32) {
            union { h16 v; h8 h[2]; } A, Bm;
            // 16-bit A 16x32 layout: lanes0-15 K{0-7,16-23}, lanes16-31 K{8-15,24-31}
            A.h[0] = *(const h8*)(pa + c0 + hi * 8);
            A.h[1] = *(const h8*)(pa + c0 + 16 + hi * 8);
            // B 32x16 layout: lanes0-15 K{0-7,16-23}, lanes16-31 K{8-15,24-31}
            Bm.h[0] = *(const h8*)(pb + c0 + hi * 8);
            Bm.h[1] = *(const h8*)(pb + c0 + 16 + hi * 8);
            acc = __builtin_amdgcn_wmma_f32_16x16x32_f16(
                false, A.v, false, Bm.v, (short)0, acc, false, false);
        }
    }

    // C/D layout: lane holds column n = lane&15; VGPR r holds row r (+8 for hi lanes)
    float* ob = out + (long)b * OH * OW;
    int n = m;                 // template row i
    int oy = yp - n + 8;
    #pragma unroll
    for (int r = 0; r < 8; ++r) {
        int ox = oxb + r + hi * 8;
        if (ox <= 96 && oy >= 0 && oy <= 96) {
            __hip_atomic_fetch_add(&ob[oy * OW + ox], acc[r] * 0.001f,
                                   __ATOMIC_RELAXED, __HIP_MEMORY_SCOPE_AGENT);
        }
    }
}

extern "C" void kernel_launch(void* const* d_in, const int* in_sizes, int n_in,
                              void* d_out, int out_size, void* d_ws, size_t ws_size,
                              hipStream_t stream) {
    const float* z  = (const float*)d_in[0];
    const float* x  = (const float*)d_in[1];
    const float* wz = (const float*)d_in[2];
    const float* bz = (const float*)d_in[3];
    const float* wx = (const float*)d_in[4];
    const float* bx = (const float*)d_in[5];
    float* out = (float*)d_out;

    // workspace: XT (100.7 MB) then ZT (2 MB)
    const size_t xt_bytes = (size_t)NB * XH * SPAD * NC * sizeof(_Float16);
    _Float16* XT = (_Float16*)d_ws;
    _Float16* ZT = (_Float16*)((char*)d_ws + xt_bytes);

    const int nout = NB * OH * OW;                      // 602176
    k_zero_out<<<(nout + 255) / 256, 256, 0, stream>>>(out, nout);

    const long nx = (long)NB * XH * SPAD * NC;          // 50,331,648
    k_prep_x<<<(unsigned)((nx + 255) / 256), 256, 0, stream>>>(x, wx, bx, XT);

    const long nz = (long)NB * ZW * ZH * NC;            // 1,048,576
    k_prep_z<<<(unsigned)((nz + 255) / 256), 256, 0, stream>>>(z, wz, bz, ZT);

    const int nwaves = NB * XH * OXT;                   // 43,008 waves
    k_corr<<<nwaves / 8, 256, 0, stream>>>(XT, ZT, out); // 8 waves / block
}
